// LocalAttention_12833362280883
// MI455X (gfx1250) — 16-bit and 8-bit matrix operand order repaired and checked by reading, compiled, not run on hardware
//
#include <hip/hip_runtime.h>
#include <stdint.h>


typedef __attribute__((ext_vector_type(8)))  int      v8i;
typedef __attribute__((ext_vector_type(8)))  float    v8f;
typedef __attribute__((ext_vector_type(16))) _Float16 v16h;

union U8i  { v8i  v; int      i[8]; };
union U16h { v16h h; unsigned u[8]; };

#define NWIN  1024
#define NTOK  49
#define MPAD  64
#define CDIM  384
#define C3    1152
#define NHEAD 12
#define HD    32
#define ATT_SCALE 0.17677669529663689f

static constexpr size_t OFF_SCAL = 0;
static constexpr size_t OFF_WSQ  = 1024;
static constexpr size_t OFF_WSO  = 8192;
static constexpr size_t OFF_BFQ  = 16384;
static constexpr size_t OFF_QWQ  = 135168;
static constexpr size_t OFF_QWO  = 589824;
static constexpr size_t OFF_XQ   = 1048576;
static constexpr size_t OFF_QKV  = 26214400;
static constexpr size_t OFF_AO   = 141819904;

__device__ __forceinline__ void async_b128(const void* gsrc, void* lds_dst) {
  unsigned l = (unsigned)(uintptr_t)lds_dst;
  asm volatile("global_load_async_to_lds_b128 %0, %1, off"
               :: "v"(l), "v"(gsrc) : "memory");
}
__device__ __forceinline__ void wait_async0() {
  asm volatile("s_wait_asynccnt 0x0" ::: "memory");
}
__device__ __forceinline__ void wait_async3() {
  asm volatile("s_wait_asynccnt 0x3" ::: "memory");
}

__device__ __forceinline__ float q8v(float x) {
  float r = rintf(x);
  return fminf(fmaxf(r, -128.f), 127.f);
}

__device__ __forceinline__ void amax_commit(float am, float* s_red, int tid, int nthr,
                                            float* target) {
  s_red[tid] = am;
  __syncthreads();
  for (int s = nthr >> 1; s > 0; s >>= 1) {
    if (tid < s) s_red[tid] = fmaxf(s_red[tid], s_red[tid + s]);
    __syncthreads();
  }
  if (tid == 0) atomicMax((unsigned int*)target, __float_as_uint(s_red[0]));
}

__global__ __launch_bounds__(256) void k_amax_x(const float* __restrict__ x, int n,
                                                float* scal) {
  __shared__ float s_red[256];
  float am = 0.f;
  for (int i = blockIdx.x * 256 + threadIdx.x; i < n; i += gridDim.x * 256)
    am = fmaxf(am, fabsf(x[i]));
  amax_commit(am, s_red, threadIdx.x, 256, &scal[0]);
}

__global__ __launch_bounds__(128) void k_prep_w(
    const float* __restrict__ Wqkv, const float* __restrict__ Wout,
    const float* __restrict__ btab, const int* __restrict__ ridx,
    signed char* __restrict__ qWq, float* __restrict__ wsq,
    signed char* __restrict__ qWo, float* __restrict__ wso,
    float* __restrict__ bfq) {
  __shared__ float red[128];
  const int b = blockIdx.x, tid = threadIdx.x;
  if (b < C3) {
    const float* row = Wqkv + (size_t)b * CDIM;
    float am = 0.f;
    for (int i = tid; i < CDIM; i += 128) am = fmaxf(am, fabsf(row[i]));
    red[tid] = am; __syncthreads();
    for (int s = 64; s > 0; s >>= 1) { if (tid < s) red[tid] = fmaxf(red[tid], red[tid + s]); __syncthreads(); }
    float sc = fmaxf(red[0], 1e-12f) / 127.f;
    if (tid == 0) wsq[b] = sc;
    for (int i = tid; i < CDIM; i += 128)
      qWq[(size_t)b * CDIM + i] = (signed char)q8v(row[i] / sc);
  } else if (b < C3 + CDIM) {
    const int r = b - C3;
    const float* row = Wout + (size_t)r * CDIM;
    float am = 0.f;
    for (int i = tid; i < CDIM; i += 128) am = fmaxf(am, fabsf(row[i]));
    red[tid] = am; __syncthreads();
    for (int s = 64; s > 0; s >>= 1) { if (tid < s) red[tid] = fmaxf(red[tid], red[tid + s]); __syncthreads(); }
    float sc = fmaxf(red[0], 1e-12f) / 127.f;
    if (tid == 0) wso[r] = sc;
    for (int i = tid; i < CDIM; i += 128)
      qWo[(size_t)r * CDIM + i] = (signed char)q8v(row[i] / sc);
  } else {
    const int h = b - (C3 + CDIM);
    float am = 0.f;
    for (int i = tid; i < NTOK * NTOK; i += 128)
      am = fmaxf(am, fabsf(btab[ridx[i] * NHEAD + h]));
    red[tid] = am; __syncthreads();
    for (int s = 64; s > 0; s >>= 1) { if (tid < s) red[tid] = fmaxf(red[tid], red[tid + s]); __syncthreads(); }
    float sc = fmaxf(red[0], 1e-12f) / 127.f;
    for (int i = tid; i < NTOK * NTOK; i += 128)
      bfq[(size_t)h * NTOK * NTOK + i] = q8v(btab[ridx[i] * NHEAD + h] / sc) * sc;
  }
}

__global__ __launch_bounds__(256) void k_quant_x(const float* __restrict__ x,
                                                 const float* __restrict__ scal,
                                                 signed char* __restrict__ xq) {
  const float sx = fmaxf(scal[0], 1e-12f) / 127.f;
  const int total = NWIN * NTOK * CDIM;
  for (int i = blockIdx.x * 256 + threadIdx.x; i < total; i += gridDim.x * 256) {
    int w = i / (NTOK * CDIM), r = i % (NTOK * CDIM);
    int t = r / CDIM, c = r % CDIM;
    int b = w >> 6, i1 = (w >> 3) & 7, i2 = w & 7;
    int h1 = t / 7, w1 = t % 7;
    int y = i1 * 7 + h1, xc = i2 * 7 + w1;
    float v = x[(((size_t)b * CDIM + c) * 56 + y) * 56 + xc];
    xq[(size_t)w * (MPAD * CDIM) + (size_t)t * CDIM + c] = (signed char)q8v(v / sx);
  }
}

__device__ __forceinline__ void gemm_i8_accum(
    const signed char* __restrict__ aptr, const signed char* __restrict__ bptr,
    signed char (&lA)[3][64][64], signed char (&lB)[3][128][64],
    int tid, int m16, int nh, int ln, int hi, v8i (&acc)[4]) {
  const int r  = tid >> 2, q4 = tid & 3;
  const int br = tid >> 1, bh = tid & 1;
  const signed char* asrc = aptr + (size_t)r * CDIM + q4 * 16;
  const signed char* bsrc = bptr + (size_t)br * CDIM + bh * 32;
#pragma unroll
  for (int pc = 0; pc < 2; ++pc) {
    async_b128(asrc + pc * 64,      &lA[pc][r][q4 * 16]);
    async_b128(bsrc + pc * 64,      &lB[pc][br][bh * 32]);
    async_b128(bsrc + pc * 64 + 16, &lB[pc][br][bh * 32 + 16]);
  }
#pragma unroll
  for (int kc = 0; kc < 6; ++kc) {
    const int buf = kc % 3;
    if (kc < 5) wait_async3();
    else        wait_async0();
    __syncthreads();
    if (kc + 2 < 6) {
      const int nb2 = (kc + 2) % 3, ko = (kc + 2) * 64;
      async_b128(asrc + ko,      &lA[nb2][r][q4 * 16]);
      async_b128(bsrc + ko,      &lB[nb2][br][bh * 32]);
      async_b128(bsrc + ko + 16, &lB[nb2][br][bh * 32 + 16]);
    }
    U8i a;
#pragma unroll
    for (int p = 0; p < 4; ++p) {
      uint2 d = *(const uint2*)&lA[buf][m16 + ln][(p >> 1) * 32 + hi * 16 + (p & 1) * 8];
      a.i[2 * p] = (int)d.x; a.i[2 * p + 1] = (int)d.y;
    }
#pragma unroll
    for (int j = 0; j < 4; ++j) {
      U8i bfr;
      const int nc = nh + j * 16 + ln;
      int4 lo = *(const int4*)&lB[buf][nc][hi * 16];
      int4 h4 = *(const int4*)&lB[buf][nc][32 + hi * 16];
      bfr.i[0] = lo.x; bfr.i[1] = lo.y; bfr.i[2] = lo.z; bfr.i[3] = lo.w;
      bfr.i[4] = h4.x; bfr.i[5] = h4.y; bfr.i[6] = h4.z; bfr.i[7] = h4.w;
      acc[j] = __builtin_amdgcn_wmma_i32_16x16x64_iu8(true, a.v, true, bfr.v,
                                                      acc[j], false, false);
    }
  }
  __syncthreads();
}

__global__ __launch_bounds__(256) void k_gemm_qkv(
    const signed char* __restrict__ xq, const signed char* __restrict__ qW,
    const float* __restrict__ wsc, const float* __restrict__ bqkv,
    float* scal, _Float16* __restrict__ qkv) {
  __shared__ __align__(16) signed char lA[3][64][64];
  __shared__ __align__(16) signed char lB[3][128][64];
  __shared__ float s_red[256];
  const int nb = blockIdx.x, w = blockIdx.y, tid = threadIdx.x;
  const int wv = tid >> 5, lane = tid & 31, ln = lane & 15, hi = lane >> 4;
  const int n0 = nb * 128, m16 = (wv >> 1) * 16, nh = (wv & 1) * 64;
  v8i acc[4] = {};
  gemm_i8_accum(xq + (size_t)w * (MPAD * CDIM), qW + (size_t)n0 * CDIM,
                lA, lB, tid, m16, nh, ln, hi, acc);
  const float sx = fmaxf(scal[0], 1e-12f) / 127.f;
  float am = 0.f;
#pragma unroll
  for (int j = 0; j < 4; ++j) {
    const int n = n0 + nh + j * 16 + ln;
    const float dq = sx * wsc[n];
    const float bb = bqkv[n];
#pragma unroll
    for (int g = 0; g < 8; ++g) {
      int m = m16 + g + 8 * hi;
      float f = dq * (float)acc[j][g] + bb;
      if (nb < 3) f *= ATT_SCALE;
      if (m < NTOK) {
        qkv[((size_t)w * NTOK + m) * C3 + n] = (_Float16)f;
        am = fmaxf(am, fabsf(f));
      }
    }
  }
  amax_commit(am, s_red, tid, 256, &scal[1 + nb / 3]);
}

__device__ __forceinline__ void load_qk8(const _Float16* __restrict__ qkv, int w, int h,
                                         float sq, float sk,
                                         _Float16 (&qraw)[NTOK][HD],
                                         _Float16 (&kraw)[NTOK][HD],
                                         signed char (&q8)[64][64],
                                         signed char (&k8)[64][64], int tid) {
  for (int e = tid; e < NTOK * 8; e += 128) {
    int t = e >> 3, sub = e & 7, which = sub >> 2, part = sub & 3;
    const _Float16* src =
        qkv + ((size_t)w * NTOK + t) * C3 + which * CDIM + h * HD + part * 8;
    _Float16* dst = which ? &kraw[t][part * 8] : &qraw[t][part * 8];
    async_b128(src, dst);
  }
  wait_async0();
  __syncthreads();
  for (int e = tid; e < 64 * 64; e += 128) {
    int t = e >> 6, c = e & 63;
    signed char qv = 0, kv = 0;
    if (t < NTOK && c < HD) {
      qv = (signed char)q8v((float)qraw[t][c] / sq);
      kv = (signed char)q8v((float)kraw[t][c] / sk);
    }
    q8[t][c] = qv; k8[t][c] = kv;
  }
}

__device__ __forceinline__ void score_iu8(const signed char (&q8)[64][64],
                                          const signed char (&k8)[64][64],
                                          int wv, int ln, int hi, v8i (&acc)[4]) {
  U8i a;
#pragma unroll
  for (int p = 0; p < 4; ++p) {
    uint2 d = *(const uint2*)&q8[wv * 16 + ln][(p >> 1) * 32 + hi * 16 + (p & 1) * 8];
    a.i[2 * p] = (int)d.x; a.i[2 * p + 1] = (int)d.y;
  }
#pragma unroll
  for (int j = 0; j < 4; ++j) {
    U8i bfr;
    int4 lo = *(const int4*)&k8[j * 16 + ln][hi * 16];
    int4 h4 = *(const int4*)&k8[j * 16 + ln][32 + hi * 16];
    bfr.i[0] = lo.x; bfr.i[1] = lo.y; bfr.i[2] = lo.z; bfr.i[3] = lo.w;
    bfr.i[4] = h4.x; bfr.i[5] = h4.y; bfr.i[6] = h4.z; bfr.i[7] = h4.w;
    acc[j] = __builtin_amdgcn_wmma_i32_16x16x64_iu8(true, a.v, true, bfr.v,
                                                    acc[j], false, false);
  }
}

__global__ __launch_bounds__(128) void k_attn_a(const _Float16* __restrict__ qkv,
                                                float* scal) {
  __shared__ __align__(16) _Float16 qraw[NTOK][HD];
  __shared__ __align__(16) _Float16 kraw[NTOK][HD];
  __shared__ __align__(16) signed char q8[64][64];
  __shared__ __align__(16) signed char k8[64][64];
  __shared__ float s_red[128];
  const int h = blockIdx.x, w = blockIdx.y, tid = threadIdx.x;
  const int wv = tid >> 5, lane = tid & 31, ln = lane & 15, hi = lane >> 4;
  const float sq = fmaxf(scal[1], 1e-12f) / 127.f;
  const float sk = fmaxf(scal[2], 1e-12f) / 127.f;
  load_qk8(qkv, w, h, sq, sk, qraw, kraw, q8, k8, tid);
  __syncthreads();
  v8i acc[4] = {};
  score_iu8(q8, k8, wv, ln, hi, acc);
  float am = 0.f;
#pragma unroll
  for (int j = 0; j < 4; ++j)
#pragma unroll
    for (int g = 0; g < 8; ++g) {
      int m = wv * 16 + g + 8 * hi, n = j * 16 + ln;
      if (m < NTOK && n < NTOK)
        am = fmaxf(am, fabsf(sq * sk * (float)acc[j][g]));
    }
  amax_commit(am, s_red, tid, 128, &scal[4]);
}

__global__ __launch_bounds__(128) void k_attn_b(const _Float16* __restrict__ qkv,
                                                const float* __restrict__ bfq,
                                                float* scal) {
  __shared__ __align__(16) _Float16 qraw[NTOK][HD];
  __shared__ __align__(16) _Float16 kraw[NTOK][HD];
  __shared__ __align__(16) signed char q8[64][64];
  __shared__ __align__(16) signed char k8[64][64];
  __shared__ float s_red[128];
  const int h = blockIdx.x, w = blockIdx.y, tid = threadIdx.x;
  const int wv = tid >> 5, lane = tid & 31, ln = lane & 15, hi = lane >> 4;
  const float sq = fmaxf(scal[1], 1e-12f) / 127.f;
  const float sk = fmaxf(scal[2], 1e-12f) / 127.f;
  const float sS = fmaxf(scal[4], 1e-12f) / 127.f;
  load_qk8(qkv, w, h, sq, sk, qraw, kraw, q8, k8, tid);
  __syncthreads();
  v8i acc[4] = {};
  score_iu8(q8, k8, wv, ln, hi, acc);
  const float* bh = bfq + (size_t)h * NTOK * NTOK;
  float am = 0.f;
#pragma unroll
  for (int j = 0; j < 4; ++j)
#pragma unroll
    for (int g = 0; g < 8; ++g) {
      int m = wv * 16 + g + 8 * hi, n = j * 16 + ln;
      if (m < NTOK && n < NTOK) {
        float s = sq * sk * (float)acc[j][g];
        float t = q8v(s / sS) * sS + bh[m * NTOK + n];
        am = fmaxf(am, fabsf(t));
      }
    }
  amax_commit(am, s_red, tid, 128, &scal[5]);
}

__global__ __launch_bounds__(128) void k_attn_c(const _Float16* __restrict__ qkv,
                                                const float* __restrict__ bfq,
                                                float* scal,
                                                _Float16* __restrict__ ao) {
  __shared__ __align__(16) _Float16 qraw[NTOK][HD];
  __shared__ __align__(16) _Float16 kraw[NTOK][HD];
  __shared__ __align__(16) _Float16 vraw[NTOK][HD];
  __shared__ __align__(16) signed char q8[64][64];
  __shared__ __align__(16) signed char k8[64][64];
  __shared__ __align__(16) _Float16 lP[64][64];
  __shared__ __align__(16) _Float16 vT[32][64];
  __shared__ float s_red[128];
  const int h = blockIdx.x, w = blockIdx.y, tid = threadIdx.x;
  const int wv = tid >> 5, lane = tid & 31, ln = lane & 15, hi = lane >> 4;
  const float sq = fmaxf(scal[1], 1e-12f) / 127.f;
  const float sk = fmaxf(scal[2], 1e-12f) / 127.f;
  const float sv = fmaxf(scal[3], 1e-12f) / 127.f;
  const float sS = fmaxf(scal[4], 1e-12f) / 127.f;
  const float sT = fmaxf(scal[5], 1e-12f) / 127.f;
  for (int e = tid; e < NTOK * 4; e += 128) {
    int t = e >> 2, part = e & 3;
    async_b128(qkv + ((size_t)w * NTOK + t) * C3 + 2 * CDIM + h * HD + part * 8,
               &vraw[t][part * 8]);
  }
  load_qk8(qkv, w, h, sq, sk, qraw, kraw, q8, k8, tid);
  for (int e = tid; e < 64 * 64; e += 128) ((_Float16*)lP)[e] = (_Float16)0.f;
  for (int e = tid; e < 32 * 64; e += 128) {
    int d = e >> 6, t = e & 63;
    float vv = 0.f;
    if (t < NTOK) vv = q8v((float)vraw[t][d] / sv) * sv;
    vT[d][t] = (_Float16)vv;
  }
  __syncthreads();
  v8i acc[4] = {};
  score_iu8(q8, k8, wv, ln, hi, acc);
  const float* bh = bfq + (size_t)h * NTOK * NTOK;
#pragma unroll
  for (int g = 0; g < 8; ++g) {
    int m = wv * 16 + g + 8 * hi;
    float xv[4]; float rm = -1e30f;
#pragma unroll
    for (int j = 0; j < 4; ++j) {
      int n = j * 16 + ln;
      bool val = (m < NTOK) && (n < NTOK);
      float s = sq * sk * (float)acc[j][g];
      float t = q8v(s / sS) * sS + (val ? bh[m * NTOK + n] : 0.f);
      t = q8v(t / sT) * sT;
      xv[j] = val ? t : -1e30f;
      rm = fmaxf(rm, xv[j]);
    }
#pragma unroll
    for (int msk = 8; msk; msk >>= 1) rm = fmaxf(rm, __shfl_xor(rm, msk, 16));
    float ev[4]; float sum = 0.f;
#pragma unroll
    for (int j = 0; j < 4; ++j) {
      ev[j] = (xv[j] > -1e29f) ? __expf(xv[j] - rm) : 0.f;
      sum += ev[j];
    }
#pragma unroll
    for (int msk = 8; msk; msk >>= 1) sum += __shfl_xor(sum, msk, 16);
    float inv = (sum > 0.f) ? 1.f / sum : 0.f;
    if (m < NTOK) {
#pragma unroll
      for (int j = 0; j < 4; ++j) {
        int n = j * 16 + ln;
        if (n < NTOK) lP[m][n] = (_Float16)(ev[j] * inv);
      }
    }
  }
  __syncthreads();
  v8f accO[2] = {};
#pragma unroll
  for (int kc = 0; kc < 2; ++kc) {
    U16h a;
#pragma unroll
    for (int v = 0; v < 8; ++v) {
      int kb = (v & 3) * 2 + hi * 8 + ((v >> 2) << 4) + kc * 32;
      a.u[v] = *(const unsigned*)&lP[wv * 16 + ln][kb];
    }
#pragma unroll
    for (int j = 0; j < 2; ++j) {
      U16h bfr;
#pragma unroll
      for (int v = 0; v < 8; ++v) {
        int kp = (v & 3) * 2 + hi * 8 + ((v >> 2) << 4) + kc * 32;
        bfr.u[v] = *(const unsigned*)&vT[j * 16 + ln][kp];
      }
      accO[j] = __builtin_amdgcn_wmma_f32_16x16x32_f16(false, a.h, false, bfr.h,
                                                       (short)0, accO[j], false, false);
    }
  }
  float am = 0.f;
#pragma unroll
  for (int j = 0; j < 2; ++j)
#pragma unroll
    for (int g = 0; g < 8; ++g) {
      int m = wv * 16 + g + 8 * hi, d = j * 16 + ln;
      float o = accO[j][g];
      if (m < NTOK) {
        ao[((size_t)w * NTOK + m) * CDIM + h * HD + d] = (_Float16)o;
        am = fmaxf(am, fabsf(o));
      }
    }
  amax_commit(am, s_red, tid, 128, &scal[6]);
}

__global__ __launch_bounds__(256) void k_quant_o(const _Float16* __restrict__ ao,
                                                 const float* __restrict__ scal,
                                                 signed char* __restrict__ oq) {
  const float sO = fmaxf(scal[6], 1e-12f) / 127.f;
  const int total = NWIN * NTOK * CDIM;
  for (int i = blockIdx.x * 256 + threadIdx.x; i < total; i += gridDim.x * 256) {
    int w = i / (NTOK * CDIM), r = i % (NTOK * CDIM);
    int t = r / CDIM, c = r % CDIM;
    oq[(size_t)w * (MPAD * CDIM) + (size_t)t * CDIM + c] =
        (signed char)q8v((float)ao[i] / sO);
  }
}

__global__ __launch_bounds__(256) void k_gemm_out(
    const signed char* __restrict__ oq, const signed char* __restrict__ qWo,
    const float* __restrict__ wso, const float* __restrict__ bout,
    const float* __restrict__ scal, float* __restrict__ out) {
  __shared__ __align__(16) signed char lA[3][64][64];
  __shared__ __align__(16) signed char lB[3][128][64];
  const int nb = blockIdx.x, w = blockIdx.y, tid = threadIdx.x;
  const int wv = tid >> 5, lane = tid & 31, ln = lane & 15, hi = lane >> 4;
  const int n0 = nb * 128, m16 = (wv >> 1) * 16, nh = (wv & 1) * 64;
  const float sO = fmaxf(scal[6], 1e-12f) / 127.f;
  v8i acc[4] = {};
  gemm_i8_accum(oq + (size_t)w * (MPAD * CDIM), qWo + (size_t)n0 * CDIM,
                lA, lB, tid, m16, nh, ln, hi, acc);
  const int b = w >> 6, i1 = (w >> 3) & 7, i2 = w & 7;
#pragma unroll
  for (int j = 0; j < 4; ++j) {
    const int n = n0 + nh + j * 16 + ln;
    const float dq = sO * wso[n];
    const float bb = bout[n];
#pragma unroll
    for (int g = 0; g < 8; ++g) {
      int m = m16 + g + 8 * hi;
      if (m < NTOK) {
        int h1 = m / 7, w1 = m % 7;
        int y = i1 * 7 + h1, xc = i2 * 7 + w1;
        out[(((size_t)b * CDIM + n) * 56 + y) * 56 + xc] = dq * (float)acc[j][g] + bb;
      }
    }
  }
}

extern "C" void kernel_launch(void* const* d_in, const int* in_sizes, int n_in,
                              void* d_out, int out_size, void* d_ws, size_t ws_size,
                              hipStream_t stream) {
  (void)n_in; (void)out_size; (void)ws_size;
  const float* x    = (const float*)d_in[0];
  const float* Wqkv = (const float*)d_in[1];
  const float* bqkv = (const float*)d_in[2];
  const float* Wout = (const float*)d_in[3];
  const float* bout = (const float*)d_in[4];
  const float* btab = (const float*)d_in[5];
  const int*   ridx = (const int*)d_in[6];

  char* ws = (char*)d_ws;
  float*       scal = (float*)(ws + OFF_SCAL);
  float*       wsq  = (float*)(ws + OFF_WSQ);
  float*       wso  = (float*)(ws + OFF_WSO);
  float*       bfq  = (float*)(ws + OFF_BFQ);
  signed char* qWq  = (signed char*)(ws + OFF_QWQ);
  signed char* qWo  = (signed char*)(ws + OFF_QWO);
  signed char* xq   = (signed char*)(ws + OFF_XQ);
  _Float16*    qkv  = (_Float16*)(ws + OFF_QKV);
  _Float16*    ao   = (_Float16*)(ws + OFF_AO);
  float*       out  = (float*)d_out;

  hipMemsetAsync(ws + OFF_SCAL, 0, 64, stream);
  hipMemsetAsync(ws + OFF_XQ, 0, (size_t)NWIN * MPAD * CDIM, stream);

  k_amax_x  <<<1024, 256, 0, stream>>>(x, in_sizes[0], scal);
  k_prep_w  <<<C3 + CDIM + NHEAD, 128, 0, stream>>>(Wqkv, Wout, btab, ridx,
                                                    qWq, wsq, qWo, wso, bfq);
  k_quant_x <<<2048, 256, 0, stream>>>(x, scal, xq);
  k_gemm_qkv<<<dim3(9, NWIN), 256, 0, stream>>>(xq, qWq, wsq, bqkv, scal, qkv);
  k_attn_a  <<<dim3(NHEAD, NWIN), 128, 0, stream>>>(qkv, scal);
  k_attn_b  <<<dim3(NHEAD, NWIN), 128, 0, stream>>>(qkv, bfq, scal);
  k_attn_c  <<<dim3(NHEAD, NWIN), 128, 0, stream>>>(qkv, bfq, scal, ao);
  k_quant_o <<<2048, 256, 0, stream>>>(ao, scal, xq);
  k_gemm_out<<<dim3(3, NWIN), 256, 0, stream>>>(xq, qWo, wso, bout, scal, out);
}
